// EncodeProcessDecodeHistory_77902116815146
// MI455X (gfx1250) — matrix operand order repaired and checked by reading, not compiled, not run on hardware
//
#include <hip/hip_runtime.h>
#include <hip/hip_bf16.h>

// ---------------------------------------------------------------------------
// EncodeProcessDecode GraphNet for MI455X (gfx1250, wave32, WMMA).
//
// Precision: f16 activations (every MLP ends in LayerNorm -> bounded range),
// f32 accumulation via V_WMMA_F32_16X16X32_F16.
// x_h (12.8 MB f16) and aggr (25.6 MB f32) are L2-resident (192 MB L2);
// e_h streams from HBM in f16 (205 MB) -> fused edge kernel is compute-bound.
//
// Tiling: workgroup = 128 threads = 4 wave32; block owns 32 rows; wave w owns
// output columns [32w, 32w+32) with FOUR 16x16 f32 accumulators
// (2 row-halves x 2 col-subtiles) so every B fragment is reused across two
// row tiles: 4 WMMAs per 4 B-loads + 4 A-loads. K loops fully unrolled.
//
// Weights are pre-packed into WMMA B-fragment-native layout:
//   Wp[kt][nt][lane][16 halves]  (contiguous 32 B per lane per fragment)
// so each B fragment is two coalesced global_load_b128.
// ---------------------------------------------------------------------------

typedef __attribute__((ext_vector_type(16))) _Float16 v16h;
typedef __attribute__((ext_vector_type(8)))  float    v8f;

static constexpr int   N_   = 50000;
static constexpr int   E_   = 800000;
static constexpr int   H_   = 128;
static constexpr int   MT_  = 32;          // rows per block
static constexpr float EPS_ = 1e-5f;

// ---------------- WMMA fragment helpers (ISA 7.12.2 layouts) ----------------

__device__ __forceinline__ v8f wmma_f16(v16h a, v16h b, v8f c) {
  return __builtin_amdgcn_wmma_f32_16x16x32_f16(
      /*neg_a=*/false, a, /*neg_b=*/false, b,
      /*c_mod=*/(short)0, c, /*reuse_a=*/false, /*reuse_b=*/false);
}

__device__ __forceinline__ v8f vzero8() {
  v8f v;
#pragma unroll
  for (int i = 0; i < 8; ++i) v[i] = 0.f;
  return v;
}

// A fragment: 16 rows x 32 K, row-major tile in LDS with given stride.
// lane&15 = row M; lanes 0-15 hold K {0..7,16..23}, lanes 16-31 hold {8..15,24..31}.
__device__ __forceinline__ v16h load_A(const _Float16* a, int stride, int koff, int lane) {
  const int m  = lane & 15;
  const int kb = (lane < 16) ? 0 : 8;
  const _Float16* r = a + m * stride + koff + kb;
  v16h v;
#pragma unroll
  for (int i = 0; i < 8; ++i) v[i] = r[i];
#pragma unroll
  for (int i = 0; i < 8; ++i) v[8 + i] = r[16 + i];
  return v;
}

// B fragment from packed weights: Wp[((kt*8+nt)*32+lane)*16 + i].
// Contiguous 32 B per lane -> two global_load_b128, coalesced across the wave.
__device__ __forceinline__ v16h load_B(const _Float16* Wp, int kt, int nt, int lane) {
  const _Float16* p = Wp + ((((size_t)kt * 8 + nt) * 32 + lane) << 4);
  v16h v;
#pragma unroll
  for (int i = 0; i < 16; ++i) v[i] = p[i];
  return v;
}

// One 32-row x 32-col GEMM step: 4 WMMAs sharing 2 B fragments.
__device__ __forceinline__ void gemm_seg2(v8f& c00, v8f& c01, v8f& c10, v8f& c11,
                                          const _Float16* aTile, int astride, int koff,
                                          const _Float16* Wp, int kt, int nt0, int lane) {
  v16h a0 = load_A(aTile, astride, koff, lane);                 // rows 0..15
  v16h a1 = load_A(aTile + 16 * astride, astride, koff, lane);  // rows 16..31
  v16h b0 = load_B(Wp, kt, nt0, lane);
  v16h b1 = load_B(Wp, kt, nt0 + 1, lane);
  c00 = wmma_f16(a0, b0, c00);
  c01 = wmma_f16(a0, b1, c01);
  c10 = wmma_f16(a1, b0, c10);
  c11 = wmma_f16(a1, b1, c11);
}

// C/D layout: c[r] -> row (lane<16 ? r : 8+r), col n0 + (lane&15).
__device__ __forceinline__ void store_half_relu_f16(_Float16* dst, const v8f& c0, const v8f& c1,
                                                    const float* bias, int n0, int lane) {
  const int nA = n0 + (lane & 15), nB = nA + 16;
  const float bA = bias[nA], bB = bias[nB];
  const int mb = (lane < 16) ? 0 : 8;
#pragma unroll
  for (int r = 0; r < 8; ++r) {
    float vA = c0[r] + bA; vA = vA > 0.f ? vA : 0.f;
    float vB = c1[r] + bB; vB = vB > 0.f ? vB : 0.f;
    dst[(mb + r) * H_ + nA] = (_Float16)vA;
    dst[(mb + r) * H_ + nB] = (_Float16)vB;
  }
}

__device__ __forceinline__ void store_half_f32(float* dst, const v8f& c0, const v8f& c1,
                                               const float* bias, int n0, int lane) {
  const int nA = n0 + (lane & 15), nB = nA + 16;
  const float bA = bias[nA], bB = bias[nB];
  const int mb = (lane < 16) ? 0 : 8;
#pragma unroll
  for (int r = 0; r < 8; ++r) {
    dst[(mb + r) * H_ + nA] = c0[r] + bA;
    dst[(mb + r) * H_ + nB] = c1[r] + bB;
  }
}

// Block-wide LayerNorm over MT x 128 f32 tile in LDS. Caller synced after fill.
__device__ __forceinline__ void block_layernorm(float* x, const float* g, const float* b,
                                                float* red, int tid) {
  if (tid < MT_) {
    const float4* r = (const float4*)(x + tid * H_);   // ds_load_b128 reduce
    float s = 0.f, s2 = 0.f;
#pragma unroll 8
    for (int i = 0; i < H_ / 4; ++i) {
      const float4 v = r[i];
      s  += v.x + v.y + v.z + v.w;
      s2 += v.x * v.x + v.y * v.y + v.z * v.z + v.w * v.w;
    }
    const float mu  = s * (1.f / 128.f);
    const float var = s2 * (1.f / 128.f) - mu * mu;
    red[tid]       = mu;
    red[MT_ + tid] = rsqrtf(var + EPS_);
  }
  __syncthreads();
  for (int i = tid; i < MT_ * H_; i += 128) {
    const int m = i >> 7, n = i & 127;
    x[i] = (x[i] - red[m]) * red[MT_ + m] * g[n] + b[n];
  }
  __syncthreads();
}

// Generic (NSEG*128 -> 128 relu -> 128 + LN) MLP on a 32-row tile.
// Segments are MT x 128 f16 tiles in LDS.
template <int NSEG>
__device__ __forceinline__ void mlp_ln_apply(const _Float16* s0, const _Float16* s1,
                                             const _Float16* s2,
                                             _Float16* sHid, float* sOut,
                                             const _Float16* W0p, const _Float16* W1p,
                                             const float* b0, const float* b1,
                                             const float* lng, const float* lnb,
                                             float* sRed, int tid) {
  const int lane = tid & 31, wave = tid >> 5, n0 = wave * 32, nt0 = wave * 2;
  const _Float16* const segs[3] = {s0, s1, s2};
  v8f c00 = vzero8(), c01 = vzero8(), c10 = vzero8(), c11 = vzero8();
#pragma unroll
  for (int t = 0; t < NSEG * 4; ++t)
    gemm_seg2(c00, c01, c10, c11, segs[t >> 2], H_, (t & 3) * 32, W0p, t, nt0, lane);
  store_half_relu_f16(sHid,            c00, c01, b0, n0, lane);
  store_half_relu_f16(sHid + 16 * H_,  c10, c11, b0, n0, lane);
  __syncthreads();
  c00 = vzero8(); c01 = vzero8(); c10 = vzero8(); c11 = vzero8();
#pragma unroll
  for (int t = 0; t < 4; ++t)
    gemm_seg2(c00, c01, c10, c11, sHid, H_, t * 32, W1p, t, nt0, lane);
  store_half_f32(sOut,           c00, c01, b1, n0, lane);
  store_half_f32(sOut + 16 * H_, c10, c11, b1, n0, lane);
  __syncthreads();
  block_layernorm(sOut, lng, lnb, sRed, tid);
}

// ------------------------------- kernels ------------------------------------

// f32 row-major W[K][128] -> f16 fragment-packed Wp (zero-padded to Kpad).
// i = ((kt*8+nt)*32+lane)*16+e ; k = kt*32 + (e<8?0:16) + (lane<16?0:8) + (e&7) ; n = nt*16+(lane&15)
__global__ void pack_w_f16_kernel(const float* __restrict__ src, _Float16* __restrict__ dst,
                                  int K, int Kpad) {
  const int i = blockIdx.x * blockDim.x + threadIdx.x;
  const int total = Kpad * H_;
  if (i >= total) return;
  const int e    = i & 15;
  const int lane = (i >> 4) & 31;
  const int nt   = (i >> 9) & 7;
  const int kt   = i >> 12;
  const int k = kt * 32 + ((e & 8) ? 16 : 0) + ((lane & 16) ? 8 : 0) + (e & 7);
  const int n = nt * 16 + (lane & 15);
  dst[i] = (k < K) ? (_Float16)src[k * H_ + n] : (_Float16)0.f;
}

__global__ void zero_f32_kernel(float* __restrict__ p, int n) {
  const int i = blockIdx.x * blockDim.x + threadIdx.x;
  if (i < n) p[i] = 0.f;
}

// Node encoder: 15 features -> 128 -> 128 + LN, f16 out. 32 nodes per block.
__global__ void __launch_bounds__(128) encode_nodes_kernel(
    const float* __restrict__ world, const float* __restrict__ pworld,
    const float* __restrict__ phi, const float* __restrict__ pphi,
    const float* __restrict__ sw, const float* __restrict__ swr,
    const float* __restrict__ swrp, const float* __restrict__ ntype,
    const float* __restrict__ nmean, const float* __restrict__ nstd,
    const _Float16* __restrict__ W0p, const _Float16* __restrict__ W1p,
    const float* __restrict__ b0, const float* __restrict__ b1,
    const float* __restrict__ lng, const float* __restrict__ lnb,
    _Float16* __restrict__ xh) {
  const int tid = threadIdx.x, lane = tid & 31, wave = tid >> 5;
  const int n0c = wave * 32, nt0 = wave * 2;
  const int node0 = blockIdx.x * MT_;
  __shared__ _Float16 sA[MT_ * 32];
  __shared__ _Float16 sHid[MT_ * H_];
  __shared__ float    sOut[MT_ * H_];
  __shared__ float    sRed[2 * MT_];

  for (int i = tid; i < MT_ * 32; i += 128) {
    const int m = i >> 5, k = i & 31;
    const int id = node0 + m;
    float f = 0.f;
    if (k < 15 && id < N_) {
      if (k < 2)       f = world[id * 2 + k] - pworld[id * 2 + k];   // u - u_prev
      else if (k == 2) f = phi[id] - pphi[id];
      else if (k == 3) f = sw[id];
      else if (k == 4) f = swr[id];
      else if (k == 5) f = swrp[id];
      else             f = ntype[id * 9 + (k - 6)];
      f = (f - nmean[k]) / nstd[k];
    }
    sA[i] = (_Float16)f;
  }
  __syncthreads();

  v8f c00 = vzero8(), c01 = vzero8(), c10 = vzero8(), c11 = vzero8();
  gemm_seg2(c00, c01, c10, c11, sA, 32, 0, W0p, 0, nt0, lane);   // K padded 15 -> 32
  store_half_relu_f16(sHid,           c00, c01, b0, n0c, lane);
  store_half_relu_f16(sHid + 16 * H_, c10, c11, b0, n0c, lane);
  __syncthreads();
  c00 = vzero8(); c01 = vzero8(); c10 = vzero8(); c11 = vzero8();
#pragma unroll
  for (int t = 0; t < 4; ++t)
    gemm_seg2(c00, c01, c10, c11, sHid, H_, t * 32, W1p, t, nt0, lane);
  store_half_f32(sOut,           c00, c01, b1, n0c, lane);
  store_half_f32(sOut + 16 * H_, c10, c11, b1, n0c, lane);
  __syncthreads();
  block_layernorm(sOut, lng, lnb, sRed, tid);
  for (int i = tid; i < MT_ * H_; i += 128)
    if (node0 + (i >> 7) < N_)
      xh[(size_t)node0 * H_ + i] = (_Float16)sOut[i];
}

// Edge encoder: 7 features -> 128 -> 128 + LN, f16 out. 32 edges per block.
__global__ void __launch_bounds__(128) encode_edges_kernel(
    const float* __restrict__ mesh, const float* __restrict__ world,
    const float* __restrict__ phi, const int* __restrict__ ei,
    const float* __restrict__ emean, const float* __restrict__ estd,
    const _Float16* __restrict__ W0p, const _Float16* __restrict__ W1p,
    const float* __restrict__ b0, const float* __restrict__ b1,
    const float* __restrict__ lng, const float* __restrict__ lnb,
    _Float16* __restrict__ eh) {
  const int tid = threadIdx.x, lane = tid & 31, wave = tid >> 5;
  const int n0c = wave * 32, nt0 = wave * 2;
  const int e0 = blockIdx.x * MT_;
  __shared__ _Float16 sA[MT_ * 32];
  __shared__ _Float16 sHid[MT_ * H_];
  __shared__ float    sOut[MT_ * H_];
  __shared__ float    sRed[2 * MT_];
  __shared__ int      sIdx[2 * MT_];  // [0..31]=row (sender), [32..63]=col (receiver)

  if (tid < MT_) { sIdx[tid] = ei[e0 + tid]; sIdx[MT_ + tid] = ei[E_ + e0 + tid]; }
  __syncthreads();

  for (int i = tid; i < MT_ * 32; i += 128) {
    const int m = i >> 5, k = i & 31;
    float f = 0.f;
    if (k < 7) {
      const int r = sIdx[m], c = sIdx[MT_ + m];
      const float dmx = mesh[r * 2]      - mesh[c * 2];
      const float dmy = mesh[r * 2 + 1]  - mesh[c * 2 + 1];
      const float dwx = world[r * 2]     - world[c * 2];
      const float dwy = world[r * 2 + 1] - world[c * 2 + 1];
      if (k == 0)      f = dmx;
      else if (k == 1) f = dmy;
      else if (k == 2) f = sqrtf(dmx * dmx + dmy * dmy);
      else if (k == 3) f = dwx;
      else if (k == 4) f = dwy;
      else if (k == 5) f = sqrtf(dwx * dwx + dwy * dwy);
      else             f = phi[r] - phi[c];
      f = (f - emean[k]) / estd[k];
    }
    sA[i] = (_Float16)f;
  }
  __syncthreads();

  v8f c00 = vzero8(), c01 = vzero8(), c10 = vzero8(), c11 = vzero8();
  gemm_seg2(c00, c01, c10, c11, sA, 32, 0, W0p, 0, nt0, lane);   // K padded 7 -> 32
  store_half_relu_f16(sHid,           c00, c01, b0, n0c, lane);
  store_half_relu_f16(sHid + 16 * H_, c10, c11, b0, n0c, lane);
  __syncthreads();
  c00 = vzero8(); c01 = vzero8(); c10 = vzero8(); c11 = vzero8();
#pragma unroll
  for (int t = 0; t < 4; ++t)
    gemm_seg2(c00, c01, c10, c11, sHid, H_, t * 32, W1p, t, nt0, lane);
  store_half_f32(sOut,           c00, c01, b1, n0c, lane);
  store_half_f32(sOut + 16 * H_, c10, c11, b1, n0c, lane);
  __syncthreads();
  block_layernorm(sOut, lng, lnb, sRed, tid);
  for (int i = tid; i < MT_ * H_; i += 128)
    eh[(size_t)e0 * H_ + i] = (_Float16)sOut[i];
}

// Fused processor edge kernel: gathers once, runs the edge MLP twice
// (message with [x_col|x_row|e], edge-update with [x_row|x_col|e]),
// atomically scatters messages into aggr, writes residual-updated e_h.
__global__ void __launch_bounds__(128) proc_edge_kernel(
    const _Float16* __restrict__ xh, _Float16* __restrict__ eh,
    float* __restrict__ aggr, const int* __restrict__ ei,
    const _Float16* __restrict__ W0p, const _Float16* __restrict__ W1p,
    const float* __restrict__ b0, const float* __restrict__ b1,
    const float* __restrict__ lng, const float* __restrict__ lnb) {
  const int tid = threadIdx.x;
  const int e0 = blockIdx.x * MT_;
  __shared__ _Float16 sXr[MT_ * H_];   // x_h[row]  (sender)
  __shared__ _Float16 sXc[MT_ * H_];   // x_h[col]  (receiver)
  __shared__ _Float16 sE[MT_ * H_];    // e_h
  __shared__ _Float16 sHid[MT_ * H_];
  __shared__ float    sOut[MT_ * H_];
  __shared__ float    sRed[2 * MT_];
  __shared__ int      sIdx[2 * MT_];

  if (tid < MT_) { sIdx[tid] = ei[e0 + tid]; sIdx[MT_ + tid] = ei[E_ + e0 + tid]; }
  // Hint the next tile of e_h toward the caches while we gather.
  if (e0 + MT_ < E_) __builtin_prefetch(eh + (size_t)(e0 + MT_) * H_, 0, 1);
  __syncthreads();

  for (int i = tid; i < MT_ * H_; i += 128) {
    const int m = i >> 7, n = i & 127;
    sXr[i] = xh[(size_t)sIdx[m] * H_ + n];        // L2-resident gathers
    sXc[i] = xh[(size_t)sIdx[MT_ + m] * H_ + n];
    sE[i]  = eh[(size_t)e0 * H_ + i];
  }
  __syncthreads();

  // message = edge_mlp([x_col | x_row | e]);  segment-sum into aggr[col]
  mlp_ln_apply<3>(sXc, sXr, sE, sHid, sOut, W0p, W1p, b0, b1, lng, lnb, sRed, tid);
  for (int i = tid; i < MT_ * H_; i += 128) {
    const int m = i >> 7, n = i & 127;
    atomicAdd(&aggr[(size_t)sIdx[MT_ + m] * H_ + n], sOut[i]);
  }
  __syncthreads();

  // new_e = edge_mlp([x_row | x_col | e]) + e   (residual), write back f16
  mlp_ln_apply<3>(sXr, sXc, sE, sHid, sOut, W0p, W1p, b0, b1, lng, lnb, sRed, tid);
  for (int i = tid; i < MT_ * H_; i += 128)
    eh[(size_t)e0 * H_ + i] = (_Float16)(sOut[i] + (float)sE[i]);
}

// Processor node kernel: x_h = node_mlp([aggr | x_h]) + x_h (in place).
__global__ void __launch_bounds__(128) proc_node_kernel(
    _Float16* __restrict__ xh, const float* __restrict__ aggr,
    const _Float16* __restrict__ W0p, const _Float16* __restrict__ W1p,
    const float* __restrict__ b0, const float* __restrict__ b1,
    const float* __restrict__ lng, const float* __restrict__ lnb) {
  const int tid = threadIdx.x;
  const int node0 = blockIdx.x * MT_;
  __shared__ _Float16 sAg[MT_ * H_];
  __shared__ _Float16 sX[MT_ * H_];
  __shared__ _Float16 sHid[MT_ * H_];
  __shared__ float    sOut[MT_ * H_];
  __shared__ float    sRed[2 * MT_];

  for (int i = tid; i < MT_ * H_; i += 128) {
    const bool ok = (node0 + (i >> 7)) < N_;
    sAg[i] = ok ? (_Float16)aggr[(size_t)node0 * H_ + i] : (_Float16)0.f;
    sX[i]  = ok ? xh[(size_t)node0 * H_ + i] : (_Float16)0.f;
  }
  __syncthreads();
  mlp_ln_apply<2>(sAg, sX, nullptr, sHid, sOut, W0p, W1p, b0, b1, lng, lnb, sRed, tid);
  for (int i = tid; i < MT_ * H_; i += 128)
    if (node0 + (i >> 7) < N_)
      xh[(size_t)node0 * H_ + i] = (_Float16)(sOut[i] + (float)sX[i]);
}

// Decoders (tiny final layers -> scalar): out[n] = [wp(2), phi(1)].
__global__ void decode_kernel(
    const _Float16* __restrict__ xh,
    const float* __restrict__ wpW0, const float* __restrict__ wpW1,
    const float* __restrict__ wpB0, const float* __restrict__ wpB1,
    const float* __restrict__ phW0, const float* __restrict__ phW1,
    const float* __restrict__ phB0, const float* __restrict__ phB1,
    float* __restrict__ out) {
  const int tid = threadIdx.x;
  const int nid = blockIdx.x;
  __shared__ float sx[H_], h1[H_], h2[H_];
  sx[tid] = (float)xh[(size_t)nid * H_ + tid];
  __syncthreads();
  float a1 = wpB0[tid], a2 = phB0[tid];
  for (int k = 0; k < H_; ++k) {
    const float xv = sx[k];
    a1 += xv * wpW0[k * H_ + tid];
    a2 += xv * phW0[k * H_ + tid];
  }
  h1[tid] = a1 > 0.f ? a1 : 0.f;
  h2[tid] = a2 > 0.f ? a2 : 0.f;
  __syncthreads();
  if (tid < 2) {
    float o = wpB1[tid];
    for (int k = 0; k < H_; ++k) o += h1[k] * wpW1[k * 2 + tid];
    out[(size_t)nid * 3 + tid] = o;
  } else if (tid == 2) {
    float o = phB1[0];
    for (int k = 0; k < H_; ++k) o += h2[k] * phW1[k];
    out[(size_t)nid * 3 + 2] = o;
  }
}

// ------------------------------ host launcher -------------------------------
//
// d_in layout (setup_inputs dict order; params leaves in construction order):
//   0 world_pos  1 prev_world_pos  2 mesh_pos  3 phi  4 prev_phi
//   5 swelling_phi  6 swelling_phi_rate  7 swelling_phi_rate_prev
//   8 node_type  9 edge_index (int, [2*E], row then col)
//   10 node_norm_mean 11 node_norm_std 12 edge_norm_mean 13 edge_norm_std
//   14..19 node_enc  {W0,W1,b0,b1,ln_g,ln_b}
//   20..25 edge_enc  {W0,W1,b0,b1,ln_g,ln_b}
//   26+12i .. 37+12i  proc i: edge_mlp{W0,W1,b0,b1,g,b}, node_mlp{W0,W1,b0,b1,g,b}
//   62..65 wp_dec {W0,W1,b0,b1}   66..69 phi_dec {W0,W1,b0,b1}

extern "C" void kernel_launch(void* const* d_in, const int* in_sizes, int n_in,
                              void* d_out, int out_size, void* d_ws, size_t ws_size,
                              hipStream_t stream) {
  auto f = [&](int i) { return (const float*)d_in[i]; };
  const float* world  = f(0);
  const float* pworld = f(1);
  const float* mesh   = f(2);
  const float* phi    = f(3);
  const float* pphi   = f(4);
  const float* sw     = f(5);
  const float* swr    = f(6);
  const float* swrp   = f(7);
  const float* ntype  = f(8);
  const int*   ei     = (const int*)d_in[9];

  // Workspace carve-up (halves for f16 pool; ~244 MB total).
  constexpr size_t W_NE0 = 0;                        // 32x128 (15 padded)
  constexpr size_t W_NE1 = 4096;                     // 128x128
  constexpr size_t W_EE0 = 20480;                    // 32x128 (7 padded)
  constexpr size_t W_EE1 = 24576;                    // 128x128
  constexpr size_t W_PROC = 40960;                   // 3 x (384+128+256+128)x128
  constexpr size_t PROC_STRIDE = 114688;
  constexpr size_t XH_OFF  = 385024;                 // halves
  constexpr size_t EH_OFF  = XH_OFF + (size_t)N_ * H_;
  constexpr size_t AG_BYTE = (EH_OFF + (size_t)E_ * H_) * 2;  // f32 aggr offset
  constexpr size_t REQ     = AG_BYTE + (size_t)N_ * H_ * 4;
  if (ws_size < REQ) return;  // workspace too small; avoid OOB

  _Float16* hp   = (_Float16*)d_ws;
  _Float16* xh   = hp + XH_OFF;
  _Float16* ehb  = hp + EH_OFF;
  float*    aggr = (float*)((char*)d_ws + AG_BYTE);

  auto pack = [&](const float* src, _Float16* dst, int K, int Kpad) {
    const int total = Kpad * H_;
    pack_w_f16_kernel<<<(total + 255) / 256, 256, 0, stream>>>(src, dst, K, Kpad);
  };

  // f32 -> f16 fragment-packed weight conversion (zero-padded K for encoders)
  pack(f(14), hp + W_NE0, 15, 32);
  pack(f(15), hp + W_NE1, 128, 128);
  pack(f(20), hp + W_EE0, 7, 32);
  pack(f(21), hp + W_EE1, 128, 128);
  _Float16 *wPE0[3], *wPE1[3], *wPN0[3], *wPN1[3];
  for (int i = 0; i < 3; ++i) {
    const size_t b = W_PROC + (size_t)i * PROC_STRIDE;
    wPE0[i] = hp + b;          wPE1[i] = hp + b + 49152;
    wPN0[i] = hp + b + 65536;  wPN1[i] = hp + b + 98304;
    const int p = 26 + i * 12;
    pack(f(p),     wPE0[i], 384, 384);
    pack(f(p + 1), wPE1[i], 128, 128);
    pack(f(p + 6), wPN0[i], 256, 256);
    pack(f(p + 7), wPN1[i], 128, 128);
  }

  const int nbNodes = (N_ + MT_ - 1) / MT_;   // 1563 (last block partially filled)
  const int nbEdges = E_ / MT_;               // 25000 (exact)

  encode_nodes_kernel<<<nbNodes, 128, 0, stream>>>(
      world, pworld, phi, pphi, sw, swr, swrp, ntype,
      f(10), f(11), hp + W_NE0, hp + W_NE1, f(16), f(17), f(18), f(19), xh);
  encode_edges_kernel<<<nbEdges, 128, 0, stream>>>(
      mesh, world, phi, ei, f(12), f(13),
      hp + W_EE0, hp + W_EE1, f(22), f(23), f(24), f(25), ehb);

  for (int i = 0; i < 3; ++i) {
    const int p = 26 + i * 12;
    zero_f32_kernel<<<(N_ * H_ + 255) / 256, 256, 0, stream>>>(aggr, N_ * H_);
    proc_edge_kernel<<<nbEdges, 128, 0, stream>>>(
        xh, ehb, aggr, ei, wPE0[i], wPE1[i], f(p + 2), f(p + 3), f(p + 4), f(p + 5));
    proc_node_kernel<<<nbNodes, 128, 0, stream>>>(
        xh, aggr, wPN0[i], wPN1[i], f(p + 8), f(p + 9), f(p + 10), f(p + 11));
  }

  decode_kernel<<<N_, 128, 0, stream>>>(
      xh, f(62), f(63), f(64), f(65), f(66), f(67), f(68), f(69), (float*)d_out);
}
